// WeaklySupervisedDetection_29205777613237
// MI455X (gfx1250) — matrix operand order repaired and checked by reading, not compiled, not run on hardware
//
#include <hip/hip_runtime.h>
#include <hip/hip_bf16.h>

typedef __attribute__((ext_vector_type(16))) __bf16 v16bf;
typedef __attribute__((ext_vector_type(8)))  __bf16 v8bf;
typedef __attribute__((ext_vector_type(2)))  __bf16 v2bf;
typedef __attribute__((ext_vector_type(8)))  float  v8f;

#define NUM_CLASSES 20
#define N_ROIS 256
#define POOL 7
#define HH 12
#define WW 12
#define CC 512
#define NOBJ (12*12*9)
#define DFC 2048

static __device__ __forceinline__ int imin(int a, int b) { return a < b ? a : b; }
static __device__ __forceinline__ int imax(int a, int b) { return a > b ? a : b; }

// ---------------------------------------------------------------------------
// 1. Top-k ROI selection (rank counting; ties -> lower index first, matching
//    jax.lax.top_k), scale + clip to integer boxes.
// ---------------------------------------------------------------------------
__global__ void topk_rois_kernel(const float* __restrict__ obj,
                                 const float* __restrict__ rois_norm,
                                 int* __restrict__ rois_i,
                                 float* __restrict__ out_rois) {
  int i = blockIdx.x * blockDim.x + threadIdx.x;
  if (i >= NOBJ) return;
  float vi = obj[i];
  int rank = 0;
  for (int j = 0; j < NOBJ; ++j) {
    float vj = obj[j];
    if (vj > vi || (vj == vi && j < i)) ++rank;
  }
  if (rank < N_ROIS) {
    float s0 = rois_norm[i * 4 + 0] * 12.0f;
    float s1 = rois_norm[i * 4 + 1] * 12.0f;
    float s2 = rois_norm[i * 4 + 2] * 12.0f;
    float s3 = rois_norm[i * 4 + 3] * 12.0f;
    int x1 = imin(imax((int)s0, 0), 11);
    int y1 = imin(imax((int)s2, 0), 11);
    int x2 = imax(imin(imax((int)s1, 0), 12), x1 + 1);
    int y2 = imax(imin(imax((int)s3, 0), 12), y1 + 1);
    rois_i[rank * 4 + 0] = x1;
    rois_i[rank * 4 + 1] = x2;
    rois_i[rank * 4 + 2] = y1;
    rois_i[rank * 4 + 3] = y2;
    out_rois[rank * 4 + 0] = (float)x1;
    out_rois[rank * 4 + 1] = (float)x2;
    out_rois[rank * 4 + 2] = (float)y1;
    out_rois[rank * 4 + 3] = (float)y2;
  }
}

// ---------------------------------------------------------------------------
// 2. ROI-align (bilinear crop-resize) -> bf16 activation matrix X (256 x 25088)
// ---------------------------------------------------------------------------
__global__ __launch_bounds__(256) void roi_align_kernel(
    const float* __restrict__ F,      // 12 x 12 x 512
    const int* __restrict__ rois,     // 256 x 4
    __bf16* __restrict__ X) {         // 256 x (7*7*512)
  const int b = blockIdx.x;
  const int x1 = rois[b * 4 + 0], x2 = rois[b * 4 + 1];
  const int y1 = rois[b * 4 + 2], y2 = rois[b * 4 + 3];
  const float hf = (float)(y2 - y1), wf = (float)(x2 - x1);

  for (int cell = 0; cell < POOL * POOL; ++cell) {
    const int py = cell / POOL, px = cell % POOL;
    float gy = ((float)py + 0.5f) / (float)POOL;
    float gx = ((float)px + 0.5f) / (float)POOL;
    float yy = (float)y1 + gy * hf - 0.5f;
    yy = fminf(fmaxf(yy, (float)y1), (float)(y2 - 1));
    float xx = (float)x1 + gx * wf - 0.5f;
    xx = fminf(fmaxf(xx, (float)x1), (float)(x2 - 1));
    int y0 = (int)floorf(yy), x0 = (int)floorf(xx);
    int y1n = imin(y0 + 1, HH - 1), x1n = imin(x0 + 1, WW - 1);
    float wy = yy - (float)y0, wx = xx - (float)x0;
    float w00 = (1.f - wy) * (1.f - wx), w01 = (1.f - wy) * wx;
    float w10 = wy * (1.f - wx), w11 = wy * wx;
    const float* p00 = F + (size_t)(y0 * WW + x0) * CC;
    const float* p01 = F + (size_t)(y0 * WW + x1n) * CC;
    const float* p10 = F + (size_t)(y1n * WW + x0) * CC;
    const float* p11 = F + (size_t)(y1n * WW + x1n) * CC;
    __bf16* dst = X + (size_t)b * (POOL * POOL * CC) + (size_t)cell * CC;
    for (int c = threadIdx.x; c < CC; c += 256) {
      float v = p00[c] * w00 + p01[c] * w01 + p10[c] * w10 + p11[c] * w11;
      dst[c] = (__bf16)v;
    }
  }
}

// ---------------------------------------------------------------------------
// 3. Fused fp32-weight -> bf16 WMMA GEMM + bias + ReLU, double-buffered LDS.
//    Y(256 x N) = relu(A(256 x K, bf16) @ W(K x N, fp32) + bias)
//    One 256-thread WG owns a 64-col strip of N and ALL 256 rows (weights
//    stream from HBM exactly once). Per 32-deep K step: next slab fetched
//    into registers (global_load_b128), 8 v_wmma_f32_16x16x32_bf16 run from
//    the current LDS buffer (all fragments = ds_load_b128, 80B row stride ->
//    bank-conflict-free), regs stored to the other buffer, ONE barrier.
//    B staging: each thread owns a 2-row x 4-col W patch so each column pair
//    packs via v_cvt_pk_bf16_f32 into a single ds_store_b32.
// ---------------------------------------------------------------------------
__global__ __launch_bounds__(256) void gemm_relu_kernel(
    const __bf16* __restrict__ A,   // 256 x K (row-major bf16)
    const float*  __restrict__ Wt,  // K x N (row-major fp32)
    const float*  __restrict__ bias,
    int K, int N,
    __bf16* __restrict__ outBf,     // 256 x N (nullable)
    float*  __restrict__ outF) {    // 256 x N (nullable)
  constexpr int AS  = 40;           // halfs per A row (80B: 16B-aligned, bank step 20)
  constexpr int BTS = 40;           // halfs per Bt row (transposed: [col][k])
  __shared__ alignas(16) __bf16 As[2][256 * AS];   // 2 x 20 KB
  __shared__ alignas(16) __bf16 Bt[2][64 * BTS];   // 2 x 5 KB

  const int tid = threadIdx.x;
  const int lane = tid & 31;
  const int wave = tid >> 5;
  const int r = lane & 15;          // col within 16-tile / row within 16-tile
  const int half = lane >> 4;       // K-half selector (wave32 WMMA layout)
  const int nstrip = wave & 3;      // which 16-col substrip
  const int mhalf = wave >> 2;      // rows 0..127 or 128..255
  const int nblk = blockIdx.x * 64;
  const int col = nblk + nstrip * 16 + r;

  const int bkr = (tid >> 4) * 2;   // 0,2,..,30: W row pair within slab
  const int bc0 = (tid & 15) * 4;   // 0..60: 4 columns within 64-strip

  union Frag { v16bf v; v8bf h[2]; };

  v8f acc[8];
#pragma unroll
  for (int m = 0; m < 8; ++m)
#pragma unroll
    for (int j = 0; j < 8; ++j) acc[m][j] = 0.0f;

  // Prologue: stage slab k0 = 0 into buffer 0.
  {
    const uint4* src = (const uint4*)(A + (size_t)tid * K);
    uint4* dst = (uint4*)&As[0][tid * AS];
    dst[0] = src[0]; dst[1] = src[1]; dst[2] = src[2]; dst[3] = src[3];
    float4 w0 = *(const float4*)(Wt + (size_t)bkr * N + nblk + bc0);
    float4 w1 = *(const float4*)(Wt + (size_t)(bkr + 1) * N + nblk + bc0);
#pragma unroll
    for (int j = 0; j < 4; ++j) {
      v2bf p;
      p[0] = (__bf16)(&w0.x)[j];
      p[1] = (__bf16)(&w1.x)[j];
      *(v2bf*)&Bt[0][(bc0 + j) * BTS + bkr] = p;
    }
  }
  __syncthreads();

  int cur = 0;
  for (int k0 = 0; k0 < K; k0 += 32, cur ^= 1) {
    const bool more = (k0 + 32) < K;
    // Fetch next slab into registers (overlaps with WMMA below).
    uint4 a0 = {}, a1 = {}, a2 = {}, a3 = {};
    float4 w0 = {}, w1 = {};
    if (more) {
      const uint4* src = (const uint4*)(A + (size_t)tid * K + (k0 + 32));
      a0 = src[0]; a1 = src[1]; a2 = src[2]; a3 = src[3];
      w0 = *(const float4*)(Wt + (size_t)(k0 + 32 + bkr) * N + nblk + bc0);
      w1 = *(const float4*)(Wt + (size_t)(k0 + 32 + bkr + 1) * N + nblk + bc0);
    }

    // B fragment: lane<16 -> K {0-7,16-23}; lane>=16 -> K {8-15,24-31} at this column.
    Frag bfrag;
    {
      const __bf16* bp = &Bt[cur][(nstrip * 16 + r) * BTS + half * 8];
      bfrag.h[0] = *(const v8bf*)(bp);
      bfrag.h[1] = *(const v8bf*)(bp + 16);
    }
    // A layout: lane<16 -> K {0-7,16-23}; lane>=16 -> K {8-15,24-31}.
    const int kofs = half * 8;
#pragma unroll
    for (int m = 0; m < 8; ++m) {
      const __bf16* ap = &As[cur][((mhalf * 8 + m) * 16 + r) * AS];
      Frag afrag;
      afrag.h[0] = *(const v8bf*)(ap + kofs);
      afrag.h[1] = *(const v8bf*)(ap + 16 + kofs);
      acc[m] = __builtin_amdgcn_wmma_f32_16x16x32_bf16(
          false, afrag.v, false, bfrag.v, (short)0, acc[m], false, false);
    }

    // Stage next slab into the other buffer.
    if (more) {
      uint4* dst = (uint4*)&As[cur ^ 1][tid * AS];
      dst[0] = a0; dst[1] = a1; dst[2] = a2; dst[3] = a3;
      __bf16* bdst = &Bt[cur ^ 1][0];
#pragma unroll
      for (int j = 0; j < 4; ++j) {
        v2bf p;
        p[0] = (__bf16)(&w0.x)[j];
        p[1] = (__bf16)(&w1.x)[j];
        *(v2bf*)&bdst[(bc0 + j) * BTS + bkr] = p;
      }
    }
    __syncthreads();
  }

  // Epilogue: C/D layout -> M = rr + half*8, N = lane%16.
  const float bv = bias ? bias[col] : 0.0f;
#pragma unroll
  for (int m = 0; m < 8; ++m) {
#pragma unroll
    for (int rr = 0; rr < 8; ++rr) {
      int grow = (mhalf * 8 + m) * 16 + rr + half * 8;
      float v = acc[m][rr] + bv;
      v = v > 0.0f ? v : 0.0f;
      size_t idx = (size_t)grow * N + col;
      if (outBf) outBf[idx] = (__bf16)v;
      if (outF)  outF[idx] = v;
    }
  }
}

// ---------------------------------------------------------------------------
// 4. Classification / detection head logits (tiny: 256 x 2048 x 20 x 2)
// ---------------------------------------------------------------------------
__global__ __launch_bounds__(256) void heads_kernel(
    const float* __restrict__ fc,
    const float* __restrict__ Wc, const float* __restrict__ bc,
    const float* __restrict__ Wd, const float* __restrict__ bd,
    float* __restrict__ logc, float* __restrict__ logd) {
  __shared__ float fs[DFC];
  const int rrow = blockIdx.x;
  for (int k = threadIdx.x; k < DFC; k += 256) fs[k] = fc[(size_t)rrow * DFC + k];
  __syncthreads();
  int o = threadIdx.x;
  if (o < NUM_CLASSES) {
    float a = 0.0f;
    for (int k = 0; k < DFC; ++k) a += fs[k] * Wc[k * NUM_CLASSES + o];
    logc[rrow * NUM_CLASSES + o] = a + bc[o];
  } else if (o < 2 * NUM_CLASSES) {
    int c = o - NUM_CLASSES;
    float a = 0.0f;
    for (int k = 0; k < DFC; ++k) a += fs[k] * Wd[k * NUM_CLASSES + c];
    logd[rrow * NUM_CLASSES + c] = a + bd[c];
  }
}

// ---------------------------------------------------------------------------
// 5. cls softmax (axis=0, over ROIs) * det softmax (axis=1, over classes),
//    scores -> d_out[20..], clipped per-class sums -> d_out[0..20)
// ---------------------------------------------------------------------------
__global__ __launch_bounds__(256) void softmax_scores_kernel(
    const float* __restrict__ logc, const float* __restrict__ logd,
    float* __restrict__ out) {
  __shared__ float red[256];
  const int t = threadIdx.x;  // ROI index
  float ld[NUM_CLASSES];
  float mx = -1e30f;
  for (int c = 0; c < NUM_CLASSES; ++c) {
    ld[c] = logd[t * NUM_CLASSES + c];
    mx = fmaxf(mx, ld[c]);
  }
  float se = 0.0f;
  for (int c = 0; c < NUM_CLASSES; ++c) { ld[c] = expf(ld[c] - mx); se += ld[c]; }
  for (int c = 0; c < NUM_CLASSES; ++c) ld[c] /= se;

  for (int c = 0; c < NUM_CLASSES; ++c) {
    float lc = logc[t * NUM_CLASSES + c];
    red[t] = lc; __syncthreads();
    for (int s = 128; s > 0; s >>= 1) {
      if (t < s) red[t] = fmaxf(red[t], red[t + s]);
      __syncthreads();
    }
    float cmax = red[0]; __syncthreads();
    float e = expf(lc - cmax);
    red[t] = e; __syncthreads();
    for (int s = 128; s > 0; s >>= 1) {
      if (t < s) red[t] += red[t + s];
      __syncthreads();
    }
    float csum = red[0]; __syncthreads();
    float sc = (e / csum) * ld[c];
    out[NUM_CLASSES + t * NUM_CLASSES + c] = sc;
    red[t] = sc; __syncthreads();
    for (int s = 128; s > 0; s >>= 1) {
      if (t < s) red[t] += red[t + s];
      __syncthreads();
    }
    if (t == 0) out[c] = fminf(fmaxf(red[0], 0.0f), 1.0f);
    __syncthreads();
  }
}

// ---------------------------------------------------------------------------
// 6. Spatial regularizer: per class, top = stable-argmin of scores column,
//    sum 0.5*((fc[r]*mask - fc[top]) * s[r])^2 over r != top, dims 0..2047.
// ---------------------------------------------------------------------------
__global__ __launch_bounds__(256) void spatial_reg_kernel(
    const float* __restrict__ scores,   // 256 x 20 (d_out + 20)
    const int* __restrict__ labels,
    const float* __restrict__ fc,       // 256 x 2048
    const int* __restrict__ rois,       // 256 x 4
    float* __restrict__ regpart) {
  __shared__ float rv[256];
  __shared__ int ri[256];
  const int c = blockIdx.x, t = threadIdx.x;
  if (labels[c] <= 0) {
    if (t == 0) regpart[c] = 0.0f;
    return;
  }
  float s = scores[t * NUM_CLASSES + c];
  rv[t] = s; ri[t] = t; __syncthreads();
  for (int st = 128; st > 0; st >>= 1) {
    if (t < st) {
      float va = rv[t], vb = rv[t + st];
      int ia = ri[t], ib = ri[t + st];
      if (vb < va || (vb == va && ib < ia)) { rv[t] = vb; ri[t] = ib; }
    }
    __syncthreads();
  }
  const int top = ri[0];
  __syncthreads();

  float rx1 = (float)rois[top * 4 + 0], rx2 = (float)rois[top * 4 + 1];
  float ry1 = (float)rois[top * 4 + 2], ry2 = (float)rois[top * 4 + 3];
  float partial = 0.0f;
  if (t != top) {
    float bx1 = (float)rois[t * 4 + 0], bx2 = (float)rois[t * 4 + 1];
    float by1 = (float)rois[t * 4 + 2], by2 = (float)rois[t * 4 + 3];
    float ix1 = fmaxf(bx1, rx1), iy1 = fmaxf(by1, ry1);
    float ix2 = fminf(bx2, rx2), iy2 = fminf(by2, ry2);
    float inter = fmaxf(ix2 - ix1 + 1.0f, 0.0f) * fmaxf(iy2 - iy1 + 1.0f, 0.0f);
    float a1 = fmaxf(bx2 - bx1 + 1.0f, 0.0f) * fmaxf(by2 - by1 + 1.0f, 0.0f);
    float a2 = fmaxf(rx2 - rx1 + 1.0f, 0.0f) * fmaxf(ry2 - ry1 + 1.0f, 0.0f);
    float iou = inter / (a1 + a2 - inter);
    float mask = (iou > 0.5f) ? 1.0f : 0.0f;
    const float* fr = fc + (size_t)t * DFC;
    const float* ft = fc + (size_t)top * DFC;
    for (int k = 0; k < DFC; ++k) {
      float d = (fr[k] * mask - ft[k]) * s;
      partial += d * d;
    }
  }
  rv[t] = partial; __syncthreads();
  for (int st = 128; st > 0; st >>= 1) {
    if (t < st) rv[t] += rv[t + st];
    __syncthreads();
  }
  if (t == 0) regpart[c] = 0.5f * rv[0];
}

__global__ void finalize_kernel(const float* __restrict__ regpart,
                                float* __restrict__ out) {
  float s = 0.0f;
  for (int c = 0; c < NUM_CLASSES; ++c) s += regpart[c];
  out[NUM_CLASSES + N_ROIS * NUM_CLASSES + N_ROIS * 4] = 0.001f * s / (float)NUM_CLASSES;
}

// ---------------------------------------------------------------------------
extern "C" void kernel_launch(void* const* d_in, const int* in_sizes, int n_in,
                              void* d_out, int out_size, void* d_ws, size_t ws_size,
                              hipStream_t stream) {
  const float* features = (const float*)d_in[0];   // 12*12*512
  const float* obj      = (const float*)d_in[1];   // 1296
  const float* roisn    = (const float*)d_in[2];   // 1296*4
  const int*   labels   = (const int*)d_in[3];     // 20
  const float* W6  = (const float*)d_in[4];
  const float* b6  = (const float*)d_in[5];
  const float* W7  = (const float*)d_in[6];
  const float* b7  = (const float*)d_in[7];
  const float* W75 = (const float*)d_in[8];
  const float* b75 = (const float*)d_in[9];
  const float* Wc  = (const float*)d_in[10];
  const float* bc  = (const float*)d_in[11];
  const float* Wd  = (const float*)d_in[12];
  const float* bd  = (const float*)d_in[13];
  float* out = (float*)d_out;

  // Workspace carve-up
  char* ws = (char*)d_ws;
  size_t off = 0;
  int*    rois_i = (int*)(ws + off);     off += (size_t)N_ROIS * 4 * sizeof(int);
  off = (off + 255) & ~(size_t)255;
  __bf16* xbf  = (__bf16*)(ws + off);    off += (size_t)N_ROIS * 25088 * 2;           // 12.8 MB
  __bf16* y1bf = (__bf16*)(ws + off);    off += (size_t)N_ROIS * 9216 * 2;            // 4.7 MB
  __bf16* y2bf = (__bf16*)(ws + off);    off += (size_t)N_ROIS * 4096 * 2;            // 2.1 MB
  float*  fcf  = (float*)(ws + off);     off += (size_t)N_ROIS * DFC * sizeof(float); // 2.1 MB
  float*  logc = (float*)(ws + off);     off += (size_t)N_ROIS * NUM_CLASSES * sizeof(float);
  float*  logd = (float*)(ws + off);     off += (size_t)N_ROIS * NUM_CLASSES * sizeof(float);
  float*  regpart = (float*)(ws + off);  off += NUM_CLASSES * sizeof(float);

  float* out_rois = out + NUM_CLASSES + N_ROIS * NUM_CLASSES;  // offset 5140
  float* scores   = out + NUM_CLASSES;                          // offset 20

  topk_rois_kernel<<<(NOBJ + 255) / 256, 256, 0, stream>>>(obj, roisn, rois_i, out_rois);
  roi_align_kernel<<<N_ROIS, 256, 0, stream>>>(features, rois_i, xbf);

  gemm_relu_kernel<<<9216 / 64, 256, 0, stream>>>(xbf, W6, b6, 25088, 9216, y1bf, nullptr);
  gemm_relu_kernel<<<4096 / 64, 256, 0, stream>>>(y1bf, W7, b7, 9216, 4096, y2bf, nullptr);
  gemm_relu_kernel<<<2048 / 64, 256, 0, stream>>>(y2bf, W75, b75, 4096, 2048, nullptr, fcf);

  heads_kernel<<<N_ROIS, 256, 0, stream>>>(fcf, Wc, bc, Wd, bd, logc, logd);
  softmax_scores_kernel<<<1, 256, 0, stream>>>(logc, logd, out);
  spatial_reg_kernel<<<NUM_CLASSES, 256, 0, stream>>>(scores, labels, fcf, rois_i, regpart);
  finalize_kernel<<<1, 1, 0, stream>>>(regpart, out);
}
